// NearestNeighborEncoder_30159260353086
// MI455X (gfx1250) — compile-verified, matrix operand order checked by reading, not run on hardware
//
#include <hip/hip_runtime.h>

// ---------------------------------------------------------------------------
// CDNA5 (gfx1250, wave32) WMMA types & helpers
// ---------------------------------------------------------------------------
typedef __attribute__((ext_vector_type(4)))  unsigned short v4u;
typedef __attribute__((ext_vector_type(8)))  unsigned short v8u;
typedef __attribute__((ext_vector_type(16))) unsigned short v16u;
typedef __attribute__((ext_vector_type(16))) __bf16        v16bf;
typedef __attribute__((ext_vector_type(8)))  float         v8f;

#define D_MODEL 1024
#define N_HEADS 16
#define D_FF_   4096
#define SEQ     2048
#define SCALE_  0.125f   // 1/sqrt(64)

__device__ __forceinline__ unsigned short f2bf(float f) {
  unsigned int u = __float_as_uint(f);
  u += 0x7FFFu + ((u >> 16) & 1u);   // round-to-nearest-even
  return (unsigned short)(u >> 16);
}

__device__ __forceinline__ v4u pack4(float a, float b, float c, float d) {
  v4u p; p[0] = f2bf(a); p[1] = f2bf(b); p[2] = f2bf(c); p[3] = f2bf(d);
  return p;
}

// 16x32 bf16 A-fragment (or B-fragment from a column-major [n][k] tile).
// ISA 7.12.2: row = lane&15; lanes 0-15 hold K=0..7 & 16..23,
// lanes 16-31 hold K=8..15 & 24..31 -> two contiguous 16B chunks per lane.
__device__ __forceinline__ v16bf load_frag(const unsigned short* base, int row,
                                           int stride, int lh) {
  const unsigned short* p = base + row * stride + lh * 8;
  v8u lo = *(const v8u*)(p);
  v8u hi = *(const v8u*)(p + 16);
  v16u t;
#pragma unroll
  for (int i = 0; i < 8; ++i) { t[i] = lo[i]; t[i + 8] = hi[i]; }
  return __builtin_bit_cast(v16bf, t);
}

__device__ __forceinline__ v8f wmma_bf16(v16bf a, v16bf b, v8f c) {
  return __builtin_amdgcn_wmma_f32_16x16x32_bf16(false, a, false, b,
                                                 (short)0, c, false, false);
}

// ---------------------------------------------------------------------------
// GEMM: C[M,N] = A[M,K] @ B[K,N] + bias (+ReLU) (+res).
// 128x128 block tile, 8 waves, each wave 32x64 (8 WMMA / k-step).
// Double-buffered LDS, float4 global loads, one barrier per k-step.
// ---------------------------------------------------------------------------
__global__ __launch_bounds__(256) void gemm_wmma(
    const float* __restrict__ A, const float* __restrict__ B,
    const float* __restrict__ bias, const float* __restrict__ res,
    float* __restrict__ C, int M, int N, int K, int relu) {
  __shared__ __align__(16) unsigned short As[2][128][40];  // [m][k]
  __shared__ __align__(16) unsigned short Bs[2][128][40];  // [n][k] col-major
  const int tid = threadIdx.x;
  const int lane = tid & 31, wave = tid >> 5;
  const int wr = wave & 3, wc = wave >> 2;       // wave tile: rows 32*wr, cols 64*wc
  const int lh = lane >> 4, l15 = lane & 15;
  const int m0 = blockIdx.x * 128, n0 = blockIdx.y * 128;

  const int arow = tid >> 3,  akq = (tid & 7) * 4;    // A: 128 rows x 8 float4
  const int bkk  = tid >> 5,  bnq = (tid & 31) * 4;   // B: 32 k-rows x 32 float4

  float4 ar[4], br[4];
  auto load_tile = [&](int k0) {
#pragma unroll
    for (int i = 0; i < 4; ++i) {
      ar[i] = *(const float4*)&A[(size_t)(m0 + arow + 32 * i) * K + k0 + akq];
      br[i] = *(const float4*)&B[(size_t)(k0 + bkk + 8 * i) * N + n0 + bnq];
    }
  };
  auto store_tile = [&](int buf) {
#pragma unroll
    for (int i = 0; i < 4; ++i) {
      *(v4u*)&As[buf][arow + 32 * i][akq] = pack4(ar[i].x, ar[i].y, ar[i].z, ar[i].w);
      int kk = bkk + 8 * i;
      Bs[buf][bnq + 0][kk] = f2bf(br[i].x);
      Bs[buf][bnq + 1][kk] = f2bf(br[i].y);
      Bs[buf][bnq + 2][kk] = f2bf(br[i].z);
      Bs[buf][bnq + 3][kk] = f2bf(br[i].w);
    }
  };

  v8f c[2][4];
#pragma unroll
  for (int i = 0; i < 2; ++i)
#pragma unroll
    for (int j = 0; j < 4; ++j) c[i][j] = (v8f){};

  load_tile(0);
  store_tile(0);
  __syncthreads();
  const int nk = K / 32;
  for (int kt = 0; kt < nk; ++kt) {
    const int buf = kt & 1;
    if (kt + 1 < nk) load_tile((kt + 1) * 32);              // hide latency
    if (kt + 2 < nk) {                                       // global_prefetch_b8
      __builtin_prefetch(&A[(size_t)(m0 + arow) * K + (kt + 2) * 32 + akq], 0, 0);
      __builtin_prefetch(&B[(size_t)((kt + 2) * 32 + bkk) * N + n0 + bnq], 0, 0);
    }
    v16bf a0 = load_frag(&As[buf][32 * wr][0],      l15, 40, lh);
    v16bf a1 = load_frag(&As[buf][32 * wr + 16][0], l15, 40, lh);
#pragma unroll
    for (int j = 0; j < 4; ++j) {
      v16bf b = load_frag(&Bs[buf][64 * wc + 16 * j][0], l15, 40, lh);
      c[0][j] = wmma_bf16(a0, b, c[0][j]);
      c[1][j] = wmma_bf16(a1, b, c[1][j]);
    }
    if (kt + 1 < nk) store_tile(buf ^ 1);
    __syncthreads();
  }
#pragma unroll
  for (int i = 0; i < 2; ++i)
#pragma unroll
    for (int j = 0; j < 4; ++j)
#pragma unroll
      for (int r = 0; r < 8; ++r) {
        int m = m0 + 32 * wr + 16 * i + r + 8 * lh;
        int n = n0 + 64 * wc + 16 * j + l15;
        float v = c[i][j][r] + bias[n];
        if (relu) v = fmaxf(v, 0.f);
        if (res)  v += res[(size_t)m * N + n];
        C[(size_t)m * N + n] = v;
      }
}

// ---------------------------------------------------------------------------
// LayerNorm, one row per workgroup
// ---------------------------------------------------------------------------
__global__ __launch_bounds__(256) void ln_kernel(const float* __restrict__ X,
                                                 const float* __restrict__ g,
                                                 const float* __restrict__ b,
                                                 float* __restrict__ Y) {
  const int row = blockIdx.x, tid = threadIdx.x;
  const float* x = X + (size_t)row * D_MODEL;
  __shared__ float red[256];
  float s = 0.f;
  for (int i = tid; i < D_MODEL; i += 256) s += x[i];
  red[tid] = s; __syncthreads();
  for (int o = 128; o > 0; o >>= 1) { if (tid < o) red[tid] += red[tid + o]; __syncthreads(); }
  float mean = red[0] * (1.f / D_MODEL);
  __syncthreads();
  float v = 0.f;
  for (int i = tid; i < D_MODEL; i += 256) { float d = x[i] - mean; v += d * d; }
  red[tid] = v; __syncthreads();
  for (int o = 128; o > 0; o >>= 1) { if (tid < o) red[tid] += red[tid + o]; __syncthreads(); }
  float inv = rsqrtf(red[0] * (1.f / D_MODEL) + 1e-5f);
  float* y = Y + (size_t)row * D_MODEL;
  for (int i = tid; i < D_MODEL; i += 256) y[i] = (x[i] - mean) * inv * g[i] + b[i];
}

// ---------------------------------------------------------------------------
// RoPE in-place on (SEQ, 16 heads, 64) viewed as (SEQ, 1024)
// ---------------------------------------------------------------------------
__global__ __launch_bounds__(256) void rope_kernel(float* __restrict__ X) {
  int idx = blockIdx.x * 256 + threadIdx.x;
  if (idx >= SEQ * N_HEADS * 32) return;
  int j = idx & 31;
  int h = (idx >> 5) & 15;
  int s = idx >> 9;
  float theta = __expf(-(float)j * (9.210340371976184f / 32.f)); // 1/10000^(j/32)
  float sn, cs;
  __sincosf((float)s * theta, &sn, &cs);
  float* p1 = X + (size_t)s * D_MODEL + h * 64 + j;
  float x1 = p1[0], x2 = p1[32];
  p1[0]  = x1 * cs - x2 * sn;
  p1[32] = x1 * sn + x2 * cs;
}

// ---------------------------------------------------------------------------
// Flash self-attention, one (head, 64-query block) per workgroup.
// Out = softmax(Q K^T * scale) V + res   (scale folded into Q)
// ---------------------------------------------------------------------------
__global__ __launch_bounds__(256) void attn_kernel(
    const float* __restrict__ Q, const float* __restrict__ Kx,
    const float* __restrict__ V, const float* __restrict__ res,
    float* __restrict__ Out) {
  const int h = blockIdx.y, qb = blockIdx.x;
  const int tid = threadIdx.x, lane = tid & 31, wave = tid >> 5;
  const int wr = wave & 3, wc = wave >> 2;
  const int lh = lane >> 4, l15 = lane & 15;
  __shared__ __align__(16) unsigned short q_s[64][72];   // [q][d] (scaled)
  __shared__ __align__(16) unsigned short k_s[64][72];   // [key][d]
  __shared__ __align__(16) unsigned short p_s[64][72];   // [q][key] bf16 probs
  __shared__ __align__(16) unsigned short vt_s[64][72];  // [d][key] (V^T)
  __shared__ float s_s[64][65];
  __shared__ float m_s[64], l_s[64], al_s[64];

  for (int i = tid; i < 64 * 16; i += 256) {             // 1024 float4s
    int r = i >> 4, dq = (i & 15) * 4;
    float4 qv = *(const float4*)&Q[(size_t)(qb * 64 + r) * D_MODEL + h * 64 + dq];
    *(v4u*)&q_s[r][dq] =
        pack4(SCALE_ * qv.x, SCALE_ * qv.y, SCALE_ * qv.z, SCALE_ * qv.w);
  }
  if (tid < 64) { m_s[tid] = -3.0e38f; l_s[tid] = 0.f; }
  v8f o0 = {}; v8f o1 = {};
  __syncthreads();

  for (int kb = 0; kb < SEQ / 64; ++kb) {
    for (int i = tid; i < 64 * 16; i += 256) {
      int key = i >> 4, dq = (i & 15) * 4;
      size_t go = (size_t)(kb * 64 + key) * D_MODEL + h * 64 + dq;
      float4 kv = *(const float4*)&Kx[go];
      float4 vv = *(const float4*)&V[go];
      *(v4u*)&k_s[key][dq] = pack4(kv.x, kv.y, kv.z, kv.w);
      vt_s[dq + 0][key] = f2bf(vv.x);
      vt_s[dq + 1][key] = f2bf(vv.y);
      vt_s[dq + 2][key] = f2bf(vv.z);
      vt_s[dq + 3][key] = f2bf(vv.w);
    }
    __syncthreads();
    v8f sc0 = {}; v8f sc1 = {};
#pragma unroll
    for (int ks = 0; ks < 64; ks += 32) {
      v16bf a  = load_frag(&q_s[16 * wr][ks],      l15, 72, lh);
      v16bf b0 = load_frag(&k_s[32 * wc][ks],      l15, 72, lh);
      v16bf b1 = load_frag(&k_s[32 * wc + 16][ks], l15, 72, lh);
      sc0 = wmma_bf16(a, b0, sc0);
      sc1 = wmma_bf16(a, b1, sc1);
    }
#pragma unroll
    for (int r = 0; r < 8; ++r) {
      int m = 16 * wr + r + 8 * lh;
      s_s[m][32 * wc + l15]      = sc0[r];
      s_s[m][32 * wc + l15 + 16] = sc1[r];
    }
    __syncthreads();
    if (tid < 64) {
      float mo = m_s[tid], mx = mo;
      for (int j = 0; j < 64; ++j) mx = fmaxf(mx, s_s[tid][j]);
      float al = __expf(mo - mx);
      float l  = l_s[tid] * al;
      for (int j = 0; j < 64; ++j) {
        float p = __expf(s_s[tid][j] - mx);
        l += p;
        p_s[tid][j] = f2bf(p);
      }
      m_s[tid] = mx; l_s[tid] = l; al_s[tid] = al;
    }
    __syncthreads();
#pragma unroll
    for (int r = 0; r < 8; ++r) {
      float al = al_s[16 * wr + r + 8 * lh];
      o0[r] *= al; o1[r] *= al;
    }
#pragma unroll
    for (int ks = 0; ks < 64; ks += 32) {
      v16bf a  = load_frag(&p_s[16 * wr][ks],       l15, 72, lh);
      v16bf b0 = load_frag(&vt_s[32 * wc][ks],      l15, 72, lh);
      v16bf b1 = load_frag(&vt_s[32 * wc + 16][ks], l15, 72, lh);
      o0 = wmma_bf16(a, b0, o0);
      o1 = wmma_bf16(a, b1, o1);
    }
    __syncthreads();
  }
#pragma unroll
  for (int r = 0; r < 8; ++r) {
    int ml = 16 * wr + r + 8 * lh;
    int mg = qb * 64 + ml;
    float inv = 1.f / l_s[ml];
    int c0i = h * 64 + 32 * wc + l15;
    size_t o = (size_t)mg * D_MODEL;
    Out[o + c0i]      = o0[r] * inv + res[o + c0i];
    Out[o + c0i + 16] = o1[r] * inv + res[o + c0i + 16];
  }
}

// ---------------------------------------------------------------------------
// Cross-attention reduced sums:
// R[h,q] = sum_k exp(L[h,q,k]) / sum_h' exp(L[h',q,k]),  L = scale * q.k
// One 32q x 16k tile per workgroup (2 waves); all 16 heads via WMMA.
// ---------------------------------------------------------------------------
__global__ __launch_bounds__(64) void cross_r_kernel(
    const float* __restrict__ Qe, const float* __restrict__ Kh,
    float* __restrict__ R) {
  const int qb = blockIdx.x, kb = blockIdx.y;
  const int tid = threadIdx.x, lane = tid & 31, wave = tid >> 5;
  const int lh = lane >> 4, l15 = lane & 15;
  __shared__ __align__(16) unsigned short q_s[32][72];
  __shared__ __align__(16) unsigned short k_s[16][72];
  __shared__ float L[N_HEADS][32][17];
  __shared__ float Racc[N_HEADS][32];
  for (int i = tid; i < N_HEADS * 32; i += 64) ((float*)Racc)[i] = 0.f;

  for (int h = 0; h < N_HEADS; ++h) {
    __syncthreads();
    for (int i = tid; i < 32 * 16; i += 64) {
      int r = i >> 4, dq = (i & 15) * 4;
      float4 qv = *(const float4*)&Qe[(size_t)(qb * 32 + r) * D_MODEL + h * 64 + dq];
      *(v4u*)&q_s[r][dq] =
          pack4(SCALE_ * qv.x, SCALE_ * qv.y, SCALE_ * qv.z, SCALE_ * qv.w);
    }
    for (int i = tid; i < 16 * 16; i += 64) {
      int r = i >> 4, dq = (i & 15) * 4;
      float4 kv = *(const float4*)&Kh[(size_t)(kb * 16 + r) * D_MODEL + h * 64 + dq];
      *(v4u*)&k_s[r][dq] = pack4(kv.x, kv.y, kv.z, kv.w);
    }
    __syncthreads();
    v8f c = {};
#pragma unroll
    for (int ks = 0; ks < 64; ks += 32) {
      v16bf a = load_frag(&q_s[16 * wave][ks], l15, 72, lh);
      v16bf b = load_frag(&k_s[0][ks],         l15, 72, lh);
      c = wmma_bf16(a, b, c);
    }
#pragma unroll
    for (int r = 0; r < 8; ++r)
      L[h][16 * wave + r + 8 * lh][l15] = c[r];
  }
  __syncthreads();

  float rl[N_HEADS];
#pragma unroll
  for (int h = 0; h < N_HEADS; ++h) rl[h] = 0.f;
  int qi = tid >> 1;
  for (int ki = (tid & 1); ki < 16; ki += 2) {
    float mx = -3.0e38f;
#pragma unroll
    for (int h = 0; h < N_HEADS; ++h) mx = fmaxf(mx, L[h][qi][ki]);
    float s = 0.f; float ex[N_HEADS];
#pragma unroll
    for (int h = 0; h < N_HEADS; ++h) { ex[h] = __expf(L[h][qi][ki] - mx); s += ex[h]; }
    float inv = 1.f / s;
#pragma unroll
    for (int h = 0; h < N_HEADS; ++h) rl[h] += ex[h] * inv;
  }
#pragma unroll
  for (int h = 0; h < N_HEADS; ++h) atomicAdd(&Racc[h][qi], rl[h]);
  __syncthreads();
  for (int i = tid; i < N_HEADS * 32; i += 64) {
    int h = i >> 5, q2 = i & 31;
    atomicAdd(&R[(size_t)h * SEQ + qb * 32 + q2], Racc[h][q2]);
  }
}

// T = V * R[h,q] + E   (elementwise, pre-Wo)
__global__ __launch_bounds__(256) void cross_scale_kernel(
    const float* __restrict__ V, const float* __restrict__ R,
    const float* __restrict__ E, float* __restrict__ T) {
  size_t idx = (size_t)blockIdx.x * 256 + threadIdx.x;
  if (idx >= (size_t)SEQ * D_MODEL) return;
  int q = (int)(idx >> 10);
  int c = (int)(idx & 1023);
  int h = c >> 6;
  T[idx] = V[idx] * R[(size_t)h * SEQ + q] + E[idx];
}

__global__ __launch_bounds__(256) void zero_kernel(float* __restrict__ p, int n) {
  int i = blockIdx.x * 256 + threadIdx.x;
  if (i < n) p[i] = 0.f;
}

// ---------------------------------------------------------------------------
// Host orchestration
// ---------------------------------------------------------------------------
static inline void launch_gemm(const float* A, const float* B, const float* bias,
                               const float* res, float* C, int M, int N, int K,
                               int relu, hipStream_t s) {
  gemm_wmma<<<dim3(M / 128, N / 128), 256, 0, s>>>(A, B, bias, res, C, M, N, K, relu);
}

extern "C" void kernel_launch(void* const* d_in, const int* in_sizes, int n_in,
                              void* d_out, int out_size, void* d_ws, size_t ws_size,
                              hipStream_t stream) {
  (void)in_sizes; (void)n_in; (void)out_size; (void)ws_size;
  const float* E_in = (const float*)d_in[0];
  const float* H_in = (const float*)d_in[1];
  int ix = 2;
  struct AttnP { const float *Wq,*bq,*Wk,*bk,*Wv,*bv,*g,*b; } ap[4];
  for (int l = 0; l < 4; ++l) {
    ap[l].Wq = (const float*)d_in[ix++]; ap[l].bq = (const float*)d_in[ix++];
    ap[l].Wk = (const float*)d_in[ix++]; ap[l].bk = (const float*)d_in[ix++];
    ap[l].Wv = (const float*)d_in[ix++]; ap[l].bv = (const float*)d_in[ix++];
    ap[l].g  = (const float*)d_in[ix++]; ap[l].b  = (const float*)d_in[ix++];
  }
  struct CaP { const float *Wq,*bq,*Wk,*bk,*Wv,*bv,*Wo,*bo,*g,*b; } cp[2];
  for (int l = 0; l < 2; ++l) {
    cp[l].Wq = (const float*)d_in[ix++]; cp[l].bq = (const float*)d_in[ix++];
    cp[l].Wk = (const float*)d_in[ix++]; cp[l].bk = (const float*)d_in[ix++];
    cp[l].Wv = (const float*)d_in[ix++]; cp[l].bv = (const float*)d_in[ix++];
    cp[l].Wo = (const float*)d_in[ix++]; cp[l].bo = (const float*)d_in[ix++];
    cp[l].g  = (const float*)d_in[ix++]; cp[l].b  = (const float*)d_in[ix++];
  }
  struct FfP { const float *W1,*b1,*W2,*b2,*g,*b; } fp[4];
  for (int l = 0; l < 4; ++l) {
    fp[l].W1 = (const float*)d_in[ix++]; fp[l].b1 = (const float*)d_in[ix++];
    fp[l].W2 = (const float*)d_in[ix++]; fp[l].b2 = (const float*)d_in[ix++];
    fp[l].g  = (const float*)d_in[ix++]; fp[l].b  = (const float*)d_in[ix++];
  }
  const float* ng = (const float*)d_in[ix++];
  const float* nb = (const float*)d_in[ix++];

  const size_t SD = (size_t)SEQ * D_MODEL;
  float* ws = (float*)d_ws;
  float* E0 = ws;
  float* E1 = E0 + SD;
  float* Nb = E1 + SD;
  float* Qb = Nb + SD;
  float* Kb = Qb + SD;
  float* Vb = Kb + SD;
  float* HN = Vb + SD;
  float* FF = HN + SD;                       // SEQ x D_FF
  float* Rb = FF + (size_t)SEQ * D_FF_;      // N_HEADS x SEQ
  float* Tb = FF;                            // alias: FF unused during cross-attn

  hipMemcpyAsync(E0, E_in, SD * sizeof(float), hipMemcpyDeviceToDevice, stream);
  ln_kernel<<<SEQ, 256, 0, stream>>>(H_in, ng, nb, HN);

  float* cur = E0; float* oth = E1;
  int ca = 0;
  const int rope_grid = (SEQ * N_HEADS * 32 + 255) / 256;
  for (int l = 0; l < 4; ++l) {
    // ---- self-attention ----
    ln_kernel<<<SEQ, 256, 0, stream>>>(cur, ap[l].g, ap[l].b, Nb);
    launch_gemm(Nb, ap[l].Wq, ap[l].bq, nullptr, Qb, SEQ, D_MODEL, D_MODEL, 0, stream);
    launch_gemm(Nb, ap[l].Wk, ap[l].bk, nullptr, Kb, SEQ, D_MODEL, D_MODEL, 0, stream);
    launch_gemm(Nb, ap[l].Wv, ap[l].bv, nullptr, Vb, SEQ, D_MODEL, D_MODEL, 0, stream);
    rope_kernel<<<rope_grid, 256, 0, stream>>>(Qb);
    rope_kernel<<<rope_grid, 256, 0, stream>>>(Kb);
    attn_kernel<<<dim3(SEQ / 64, N_HEADS), 256, 0, stream>>>(Qb, Kb, Vb, cur, oth);
    { float* t = cur; cur = oth; oth = t; }

    // ---- cross-attention (layers 1,3) ----
    if (l == 1 || l == 3) {
      const CaP& c = cp[ca++];
      ln_kernel<<<SEQ, 256, 0, stream>>>(cur, c.g, c.b, Nb);
      launch_gemm(Nb, c.Wq, c.bq, nullptr, Qb, SEQ, D_MODEL, D_MODEL, 0, stream);
      launch_gemm(HN, c.Wk, c.bk, nullptr, Kb, SEQ, D_MODEL, D_MODEL, 0, stream);
      launch_gemm(HN, c.Wv, c.bv, nullptr, Vb, SEQ, D_MODEL, D_MODEL, 0, stream);
      zero_kernel<<<(N_HEADS * SEQ + 255) / 256, 256, 0, stream>>>(Rb, N_HEADS * SEQ);
      cross_r_kernel<<<dim3(SEQ / 32, SEQ / 16), 64, 0, stream>>>(Qb, Kb, Rb);
      cross_scale_kernel<<<(int)((SD + 255) / 256), 256, 0, stream>>>(Vb, Rb, cur, Tb);
      launch_gemm(Tb, c.Wo, c.bo, nullptr, oth, SEQ, D_MODEL, D_MODEL, 0, stream);
      { float* t = cur; cur = oth; oth = t; }
    }

    // ---- feed-forward ----
    ln_kernel<<<SEQ, 256, 0, stream>>>(cur, fp[l].g, fp[l].b, Nb);
    launch_gemm(Nb, fp[l].W1, fp[l].b1, nullptr, FF, SEQ, D_FF_, D_MODEL, 1, stream);
    launch_gemm(FF, fp[l].W2, fp[l].b2, cur, oth, SEQ, D_MODEL, D_FF_, 0, stream);
    { float* t = cur; cur = oth; oth = t; }
  }
  hipMemcpyAsync(d_out, cur, SD * sizeof(float), hipMemcpyDeviceToDevice, stream);
}
